// RoleDecoder_17437567222269
// MI455X (gfx1250) — matrix operand order repaired and checked by reading, not compiled, not run on hardware
//
#include <hip/hip_runtime.h>
#include <math.h>

// Problem constants (match reference)
#define R_ 8
#define B_ 16
#define S_ 512
#define H_ 768
#define E_ 32
#define M_ (B_ * S_)        // 8192 rows
#define KB_ (H_ / 32)       // 24 K-blocks of 32
#define NT_ (H_ / 16)       // 48 N-tiles of 16
#define RG_ 4               // 16-row groups per block -> 64-row M-tile
#define EPS_ 1e-7f

typedef __attribute__((ext_vector_type(16))) _Float16 v16h;
typedef __attribute__((ext_vector_type(8)))  float    v8f;

// ---------------------------------------------------------------------------
// Fragment-order packing (wave32 WMMA layouts per ISA 7.12.2).
//  A 16x32 f16 (MxK): lane<16 row=lane, half pairs v=0..3 -> K=0,1..6,7 ;
//  v=4..7 -> K=16..23.  lane>=16: same rows, K offset +8.
//  B 32x16 f16 (KxN): lane%16 = N col; same K order as A (h<8 -> K=h, h>=8 -> K=16+h-8; lane>=16 +8).
//  C/D 16x16 f32: col = lane%16, row = v + 8*(lane>=16).
//
// Packed A:   [strip16][kb][lane][16 halves]  -> 32B contiguous per lane
// Packed B:   [kb][ntile][lane][16 halves]    -> 32B contiguous per lane
// Packed C:   [strip16][ntile][lane][8 f32]   -> 32B contiguous per lane
// ---------------------------------------------------------------------------

__global__ void __launch_bounds__(256) pack_a_kernel(const float* __restrict__ A,
                                                     _Float16* __restrict__ out,
                                                     int nstrips) {
  int tid = blockIdx.x * 256 + threadIdx.x;        // (strip*KB_ + kb)*32 + lane
  if (tid >= nstrips * KB_ * 32) return;
  int lane  = tid & 31;
  int kb    = (tid >> 5) % KB_;
  int strip = tid / (KB_ * 32);
  int row = strip * 16 + (lane & 15);
  int kh  = (lane & 16) ? 8 : 0;
  const float* p = A + (size_t)row * H_ + kb * 32;
  _Float16* o = out + (size_t)tid * 16;
#pragma unroll
  for (int h = 0; h < 16; ++h) {
    int v = h >> 1, j = h & 1;
    int k = ((v & 4) ? 16 : 0) + kh + (v & 3) * 2 + j;
    o[h] = (_Float16)p[k];
  }
}

__global__ void __launch_bounds__(256) pack_b_kernel(const float* __restrict__ W,
                                                     _Float16* __restrict__ out) {
  int tid = blockIdx.x * 256 + threadIdx.x;        // (kb*NT_ + nt)*32 + lane
  if (tid >= KB_ * NT_ * 32) return;
  int lane = tid & 31;
  int nt   = (tid >> 5) % NT_;
  int kb   = tid / (NT_ * 32);
  int col   = nt * 16 + (lane & 15);
  int kbase = kb * 32 + ((lane & 16) ? 8 : 0);
  _Float16* o = out + (size_t)tid * 16;
#pragma unroll
  for (int h = 0; h < 16; ++h)
    o[h] = (_Float16)W[(size_t)(kbase + ((h & 8) ? 16 : 0) + (h & 7)) * H_ + col];
}

// ---------------------------------------------------------------------------
// 64-row x 96-col accumulate: four A row-groups share each B fragment, so one
// B load-wait amortizes over 4 WMMAs and each block reads the 1.1 MB packed W
// slice once per 64 output rows (halving L2 refetch vs 32-row tiles).
// ---------------------------------------------------------------------------
__device__ __forceinline__ void wmma_accum_strip64(
    const _Float16* __restrict__ Apk, const _Float16* __restrict__ Wpk,
    int gbase, int wave, int lane, v8f (&acc)[RG_][6]) {
  const int nt0 = wave * 6;
  for (int kb = 0; kb < KB_; ++kb) {
    v16h a[RG_];
#pragma unroll
    for (int r = 0; r < RG_; ++r)
      a[r] = *(const v16h*)(Apk + (((size_t)(gbase + r) * KB_ + kb) * 32 + lane) * 16);
#pragma unroll
    for (int t = 0; t < 6; ++t) {
      v16h b = *(const v16h*)(Wpk + (((size_t)kb * NT_ + nt0 + t) * 32 + lane) * 16);
#pragma unroll
      for (int r = 0; r < RG_; ++r)
        acc[r][t] = __builtin_amdgcn_wmma_f32_16x16x32_f16(false, a[r], false, b,
                                                           (short)0, acc[r][t], false, false);
    }
  }
}

// ---------------------------------------------------------------------------
// base = A1 @ W1 + A2 @ W2 + bias  (role-invariant precompute, C-fragment f32)
// grid.x = M/64 tiles, 256 threads = 8 waves, each wave: 4 row-groups x 6 tiles.
// ---------------------------------------------------------------------------
__global__ void __launch_bounds__(256) base_gemm_kernel(
    const _Float16* __restrict__ A1, const _Float16* __restrict__ W1,
    const _Float16* __restrict__ A2, const _Float16* __restrict__ W2,
    const float* __restrict__ bias, float* __restrict__ Cpk) {
  int lane = threadIdx.x & 31;
  int wave = threadIdx.x >> 5;
  int gbase = blockIdx.x * RG_;

  v8f z = {};
  v8f acc[RG_][6];
#pragma unroll
  for (int r = 0; r < RG_; ++r)
#pragma unroll
    for (int t = 0; t < 6; ++t) acc[r][t] = z;

  wmma_accum_strip64(A1, W1, gbase, wave, lane, acc);
  wmma_accum_strip64(A2, W2, gbase, wave, lane, acc);

#pragma unroll
  for (int t = 0; t < 6; ++t) {
    int nt = wave * 6 + t;
    float bv = bias[nt * 16 + (lane & 15)];
#pragma unroll
    for (int r = 0; r < RG_; ++r) {
      v8f rr = acc[r][t];
#pragma unroll
      for (int v = 0; v < 8; ++v) rr[v] += bv;
      *(v8f*)(Cpk + (((size_t)(gbase + r) * NT_ + nt) * 32 + lane) * 8) = rr;
    }
  }
}

// ---------------------------------------------------------------------------
// Fused per-role MLP head:
//   h = base + role_i @ Wmid + pre_answer*w_last      (WMMA GEMM + epilogue)
//   out = sigmoid( sum_col( gelu(h) * W2[col] ) + b2 )  -> [M] scores
// ---------------------------------------------------------------------------
__global__ void __launch_bounds__(256) mlp_head_kernel(
    const _Float16* __restrict__ Apk, const _Float16* __restrict__ Wpk,
    const float* __restrict__ basepk, const float* __restrict__ w_last,
    const float* __restrict__ pa,     const float* __restrict__ W2,
    const float* __restrict__ b2,     float* __restrict__ out) {
  __shared__ float red[16 * RG_];
  int lane = threadIdx.x & 31;
  int wave = threadIdx.x >> 5;
  int gbase = blockIdx.x * RG_;

  if (threadIdx.x < 16 * RG_) red[threadIdx.x] = 0.0f;
  __syncthreads();

  v8f z = {};
  v8f acc[RG_][6];
#pragma unroll
  for (int r = 0; r < RG_; ++r)
#pragma unroll
    for (int t = 0; t < 6; ++t) acc[r][t] = z;

  wmma_accum_strip64(Apk, Wpk, gbase, wave, lane, acc);

  int off = (lane & 16) ? 8 : 0;
  v8f pav[RG_];
#pragma unroll
  for (int r = 0; r < RG_; ++r)
    pav[r] = *(const v8f*)(pa + (gbase + r) * 16 + off);

  const float kInvSqrt2 = 0.7071067811865475f;
  float psum[RG_][8];
#pragma unroll
  for (int r = 0; r < RG_; ++r)
#pragma unroll
    for (int v = 0; v < 8; ++v) psum[r][v] = 0.0f;

#pragma unroll
  for (int t = 0; t < 6; ++t) {
    int nt = wave * 6 + t;
    int col = nt * 16 + (lane & 15);
    float w2c = W2[col];
    float wlc = w_last[col];
#pragma unroll
    for (int r = 0; r < RG_; ++r) {
      v8f bb = *(const v8f*)(basepk + (((size_t)(gbase + r) * NT_ + nt) * 32 + lane) * 8);
#pragma unroll
      for (int v = 0; v < 8; ++v) {
        float h = acc[r][t][v] + bb[v] + pav[r][v] * wlc;
        float g = 0.5f * h * (1.0f + erff(h * kInvSqrt2));   // exact GELU
        psum[r][v] += g * w2c;
      }
    }
  }
#pragma unroll
  for (int r = 0; r < RG_; ++r)
#pragma unroll
    for (int v = 0; v < 8; ++v)
      atomicAdd(&red[r * 16 + off + v], psum[r][v]);
  __syncthreads();

  if (threadIdx.x < 16 * RG_) {
    float x = red[threadIdx.x] + b2[0];
    out[gbase * 16 + threadIdx.x] = 1.0f / (1.0f + expf(-x));
  }
}

// ---------------------------------------------------------------------------
// char_scores[b,e] = max over entity_token_spans of multi[b, :]
// ---------------------------------------------------------------------------
__global__ void __launch_bounds__(256) span_entity_kernel(
    const float* __restrict__ multi, const int* __restrict__ spans,
    float* __restrict__ cs) {
  int idx = blockIdx.x * blockDim.x + threadIdx.x;
  if (idx >= B_ * E_) return;
  int b = idx / E_;
  int st = spans[2 * idx], en = spans[2 * idx + 1];
  float m = -INFINITY;
  for (int t = st; t < en; ++t) m = fmaxf(m, multi[b * S_ + t]);
  cs[idx] = m;
}

// ---------------------------------------------------------------------------
// pred = 0.5*(span_max(single, char2token) + scatter(char_scores, doc_spans))
// + BCE partial sum -> loss_acc
// ---------------------------------------------------------------------------
__global__ void __launch_bounds__(256) pred_kernel(
    const float* __restrict__ single, const int* __restrict__ c2t,
    const float* __restrict__ cs, const int* __restrict__ dspans,
    const float* __restrict__ labels, float* __restrict__ pred_out,
    float* __restrict__ loss_acc) {
  __shared__ float sh[256];
  int idx = blockIdx.x * 256 + threadIdx.x;   // < M_
  int b = idx >> 9;
  int s = idx & (S_ - 1);

  int st = c2t[2 * idx], en = c2t[2 * idx + 1];
  float m = -INFINITY;
  for (int t = st; t < en; ++t) m = fmaxf(m, single[b * S_ + t]);

  float pe = 0.0f;
  for (int e = 0; e < E_; ++e) {
    int ds = dspans[2 * (b * E_ + e)];
    int de = dspans[2 * (b * E_ + e) + 1];
    if (s >= ds && s < de) pe += cs[b * E_ + e];
  }

  float pr = 0.5f * (m + pe);
  pred_out[idx] = pr;

  float p = fminf(fmaxf(pr, EPS_), 1.0f - EPS_);
  float y = labels[idx];
  float bce = -(y * logf(p) + (1.0f - y) * logf(1.0f - p));

  sh[threadIdx.x] = bce;
  __syncthreads();
  for (int stp = 128; stp > 0; stp >>= 1) {
    if (threadIdx.x < stp) sh[threadIdx.x] += sh[threadIdx.x + stp];
    __syncthreads();
  }
  if (threadIdx.x == 0) atomicAdd(loss_acc, sh[0]);
}

// ---------------------------------------------------------------------------
// pre_answer' = concat([pred, pre_answer]) @ Wa + ba   (Wa: [2S, S])
// ---------------------------------------------------------------------------
__global__ void __launch_bounds__(256) preanswer_kernel(
    const float* __restrict__ pred, const float* __restrict__ pa_old,
    const float* __restrict__ Wa, const float* __restrict__ ba,
    float* __restrict__ pa_new) {
  int idx = blockIdx.x * 256 + threadIdx.x;   // < M_
  int b = idx >> 9;
  int s = idx & (S_ - 1);
  const float* pr = pred + b * S_;
  const float* po = pa_old + b * S_;
  float acc = ba[s];
  for (int j = 0; j < S_; ++j) acc += pr[j] * Wa[(size_t)j * S_ + s];
  for (int j = 0; j < S_; ++j) acc += po[j] * Wa[(size_t)(S_ + j) * S_ + s];
  pa_new[idx] = acc;
}

__global__ void init_kernel(float* __restrict__ pa, float* __restrict__ loss_acc) {
  int i = blockIdx.x * blockDim.x + threadIdx.x;
  if (i < M_) pa[i] = 0.0f;
  if (i == 0) *loss_acc = 0.0f;
}

__global__ void __launch_bounds__(256) mask_sum_kernel(const int* __restrict__ mask,
                                                       float* __restrict__ msum) {
  __shared__ float sh[256];
  float s = 0.0f;
  for (int i = threadIdx.x; i < M_; i += 256) s += (float)mask[i];
  sh[threadIdx.x] = s;
  __syncthreads();
  for (int stp = 128; stp > 0; stp >>= 1) {
    if (threadIdx.x < stp) sh[threadIdx.x] += sh[threadIdx.x + stp];
    __syncthreads();
  }
  if (threadIdx.x == 0) *msum = sh[0];
}

// total_loss = sum_i bce_mean_i * mask_sum = (loss_acc / (B*S)) * mask_sum
__global__ void finalize_kernel(const float* __restrict__ loss_acc,
                                const float* __restrict__ msum,
                                float* __restrict__ out) {
  out[0] = (*loss_acc) * (*msum) / (float)M_;
}

// ---------------------------------------------------------------------------
extern "C" void kernel_launch(void* const* d_in, const int* in_sizes, int n_in,
                              void* d_out, int out_size, void* d_ws, size_t ws_size,
                              hipStream_t stream) {
  const float* role   = (const float*)d_in[0];   // [R,B,S,H]
  const float* labels = (const float*)d_in[1];   // [R,B,S]
  const float* summar = (const float*)d_in[2];   // [B,S,H]
  const float* token  = (const float*)d_in[3];   // [B,S,H]
  const float* ents   = (const float*)d_in[4];   // [B,S,H]
  const int*   mask   = (const int*)d_in[5];     // [B,S]
  const int*   c2t    = (const int*)d_in[6];     // [B,S,2]
  const int*   etok   = (const int*)d_in[7];     // [B,E,2]
  const int*   edoc   = (const int*)d_in[8];     // [B,E,2]
  const float* W1s = (const float*)d_in[9];      // [3H+1, H]
  const float* b1s = (const float*)d_in[10];
  const float* W2s = (const float*)d_in[11];     // [H,1]
  const float* b2s = (const float*)d_in[12];
  const float* W1m = (const float*)d_in[13];
  const float* b1m = (const float*)d_in[14];
  const float* W2m = (const float*)d_in[15];
  const float* b2m = (const float*)d_in[16];
  const float* Wa  = (const float*)d_in[17];     // [2S, S]
  const float* ba  = (const float*)d_in[18];

  float* out = (float*)d_out;                    // [1 + R*B*S]

  // ---- workspace carve-up (~196 MB) ----
  char* wptr = (char*)d_ws;
  auto alloc = [&](size_t bytes) -> char* {
    char* p = wptr;
    wptr += (bytes + 255) & ~(size_t)255;
    return p;
  };
  _Float16* role_pk = (_Float16*)alloc((size_t)R_ * M_ * H_ * 2);
  _Float16* sum_pk  = (_Float16*)alloc((size_t)M_ * H_ * 2);
  _Float16* tok_pk  = (_Float16*)alloc((size_t)M_ * H_ * 2);
  _Float16* ent_pk  = (_Float16*)alloc((size_t)M_ * H_ * 2);
  _Float16* ws_sum_pk = (_Float16*)alloc((size_t)H_ * H_ * 2);
  _Float16* ws_mid_pk = (_Float16*)alloc((size_t)H_ * H_ * 2);
  _Float16* ws_tok_pk = (_Float16*)alloc((size_t)H_ * H_ * 2);
  _Float16* wm_sum_pk = (_Float16*)alloc((size_t)H_ * H_ * 2);
  _Float16* wm_mid_pk = (_Float16*)alloc((size_t)H_ * H_ * 2);
  _Float16* wm_ent_pk = (_Float16*)alloc((size_t)H_ * H_ * 2);
  float* base_s = (float*)alloc((size_t)M_ * H_ * 4);
  float* base_m = (float*)alloc((size_t)M_ * H_ * 4);
  float* single_b = (float*)alloc(M_ * 4);
  float* multi_b  = (float*)alloc(M_ * 4);
  float* pa0      = (float*)alloc(M_ * 4);
  float* pa1      = (float*)alloc(M_ * 4);
  float* cscore   = (float*)alloc(B_ * E_ * 4);
  float* loss_acc = (float*)alloc(4);
  float* msum     = (float*)alloc(4);

  // Weight slices of W1 (row-major [3H+1, H])
  const float* ws_last = W1s + (size_t)3 * H_ * H_;  // row 3H
  const float* wm_last = W1m + (size_t)3 * H_ * H_;

  init_kernel<<<(M_ + 255) / 256, 256, 0, stream>>>(pa0, loss_acc);
  mask_sum_kernel<<<1, 256, 0, stream>>>(mask, msum);

  // ---- pack activations + weights into f16 fragment order (once) ----
  {
    int strips_role = R_ * M_ / 16;
    int th_role = strips_role * KB_ * 32;
    pack_a_kernel<<<(th_role + 255) / 256, 256, 0, stream>>>(role, role_pk, strips_role);
    int strips = M_ / 16;
    int th = strips * KB_ * 32;
    pack_a_kernel<<<(th + 255) / 256, 256, 0, stream>>>(summar, sum_pk, strips);
    pack_a_kernel<<<(th + 255) / 256, 256, 0, stream>>>(token,  tok_pk, strips);
    pack_a_kernel<<<(th + 255) / 256, 256, 0, stream>>>(ents,   ent_pk, strips);

    int bth = KB_ * NT_ * 32;
    int bblk = (bth + 255) / 256;
    pack_b_kernel<<<bblk, 256, 0, stream>>>(W1s,                         ws_sum_pk);
    pack_b_kernel<<<bblk, 256, 0, stream>>>(W1s + (size_t)H_ * H_,       ws_mid_pk);
    pack_b_kernel<<<bblk, 256, 0, stream>>>(W1s + (size_t)2 * H_ * H_,   ws_tok_pk);
    pack_b_kernel<<<bblk, 256, 0, stream>>>(W1m,                         wm_sum_pk);
    pack_b_kernel<<<bblk, 256, 0, stream>>>(W1m + (size_t)H_ * H_,       wm_mid_pk);
    pack_b_kernel<<<bblk, 256, 0, stream>>>(W1m + (size_t)2 * H_ * H_,   wm_ent_pk);
  }

  // ---- role-invariant bases (reused by all 8 roles) ----
  base_gemm_kernel<<<M_ / (16 * RG_), 256, 0, stream>>>(sum_pk, ws_sum_pk, tok_pk,
                                                        ws_tok_pk, b1s, base_s);
  base_gemm_kernel<<<M_ / (16 * RG_), 256, 0, stream>>>(sum_pk, wm_sum_pk, ent_pk,
                                                        wm_ent_pk, b1m, base_m);

  // ---- sequential role loop (pre_answer recurrence) ----
  float* pa_cur = pa0;
  float* pa_nxt = pa1;
  for (int i = 0; i < R_; ++i) {
    const _Float16* re_pk = role_pk + (size_t)i * M_ * H_;
    const float* lab = labels + (size_t)i * M_;
    float* pred_i = out + 1 + (size_t)i * M_;

    mlp_head_kernel<<<M_ / (16 * RG_), 256, 0, stream>>>(re_pk, ws_mid_pk, base_s,
                                                         ws_last, pa_cur, W2s, b2s,
                                                         single_b);
    mlp_head_kernel<<<M_ / (16 * RG_), 256, 0, stream>>>(re_pk, wm_mid_pk, base_m,
                                                         wm_last, pa_cur, W2m, b2m,
                                                         multi_b);
    span_entity_kernel<<<(B_ * E_ + 255) / 256, 256, 0, stream>>>(multi_b, etok, cscore);
    pred_kernel<<<M_ / 256, 256, 0, stream>>>(single_b, c2t, cscore, edoc, lab,
                                              pred_i, loss_acc);
    preanswer_kernel<<<M_ / 256, 256, 0, stream>>>(pred_i, pa_cur, Wa, ba, pa_nxt);

    float* t = pa_cur; pa_cur = pa_nxt; pa_nxt = t;
  }

  finalize_kernel<<<1, 1, 0, stream>>>(loss_acc, msum, out);
}
